// LIIF_6949257084959
// MI455X (gfx1250) — matrix operand order repaired and checked by reading, not compiled, not run on hardware
//
#include <hip/hip_runtime.h>
#include <hip/hip_bf16.h>

// ---------------- problem constants ----------------
#define LIIF_B   8
#define LIIF_Q   8192
#define LIIF_BQ  (LIIF_B * LIIF_Q)     // 65536 samples
#define LIIF_HW  64
#define LIIF_C   64
#define LIIF_HID 256
#define LIIF_KIN 608                   // 580 padded up to 19*32
#define LIIF_MT  64                    // rows (samples*corners) per workgroup

typedef __attribute__((ext_vector_type(16))) _Float16     v16h;
typedef __attribute__((ext_vector_type(8)))  float        v8f;
typedef __attribute__((ext_vector_type(4)))  unsigned int u32x4;

union Frag { v16h h; u32x4 u[2]; };

// ---------------- conv 3x3 encoder: inp(8,3,64,64) -> featH(8,64,64,64) f16 HWC ----------------
__global__ __launch_bounds__(256) void liif_conv(
    const float* __restrict__ inp, const float* __restrict__ ew,
    const float* __restrict__ eb, _Float16* __restrict__ featH)
{
    int idx = blockIdx.x * 256 + threadIdx.x;           // (b,y,x,c) c-fastest
    if (idx >= LIIF_B * LIIF_HW * LIIF_HW * LIIF_C) return;
    int c = idx & 63;
    int x = (idx >> 6) & 63;
    int y = (idx >> 12) & 63;
    int b = idx >> 18;
    float s = eb[c];
    #pragma unroll
    for (int ci = 0; ci < 3; ++ci)
        #pragma unroll
        for (int ky = 0; ky < 3; ++ky) {
            int yy = y + ky - 1;
            if (yy < 0 || yy > 63) continue;
            #pragma unroll
            for (int kx = 0; kx < 3; ++kx) {
                int xx = x + kx - 1;
                if (xx < 0 || xx > 63) continue;
                s += inp[((b * 3 + ci) * 64 + yy) * 64 + xx] *
                     ew[((c * 3 + ci) * 3 + ky) * 3 + kx];
            }
        }
    featH[idx] = (_Float16)s;
}

// ---------------- weight prep: transpose + f16 + K-permutation for layer0 ----------------
__global__ __launch_bounds__(256) void liif_prep_wt0(
    const float* __restrict__ w0, _Float16* __restrict__ Wt0)
{
    int idx = blockIdx.x * 256 + threadIdx.x;           // 256 * 608
    if (idx >= LIIF_HID * LIIF_KIN) return;
    int n = idx / LIIF_KIN, k = idx % LIIF_KIN;
    float v = 0.f;
    if (k < 576) {                       // k' = p*64 + c  <->  k_orig = c*9 + p
        int p = k >> 6, c = k & 63;
        v = w0[(c * 9 + p) * LIIF_HID + n];
    } else if (k < 580) {
        v = w0[k * LIIF_HID + n];
    }
    Wt0[idx] = (_Float16)v;
}

__global__ __launch_bounds__(256) void liif_prep_wth(
    const float* __restrict__ w, _Float16* __restrict__ Wt)
{
    int idx = blockIdx.x * 256 + threadIdx.x;           // 256 * 256
    int n = idx >> 8, k = idx & 255;
    Wt[idx] = (_Float16)w[k * LIIF_HID + n];            // Wt[n][k] = w[k][n]
}

// ---------------- fused WMMA MLP layer (wave32) ----------------
// 8 waves: wave w owns columns [32w, 32w+32); iterates 4 M-tiles of 16 rows.
// A (16x32 f16): lane row = lane&15; halves 0-15 -> K = ks + (lane<16?0:16) + h
//                (two 16B LDS loads).
// B (32x16 f16): lane col = lane&15; halves 0-15 -> K = ks + (lane<16?0:16) + h
//                (two 16B global loads from row-major Wt[n][K]).
__device__ __forceinline__ void mlp_layer(
    const _Float16* src, int srcStride, int K,
    const _Float16* __restrict__ Wt, const float* __restrict__ bias,
    _Float16* dst, int dstStride, bool relu, int lane, int wave)
{
    const int hsel = lane >> 4;          // 0: lanes 0-15, 1: lanes 16-31
    const int l16  = lane & 15;
    const int n0 = wave * 32 + l16;
    const int n1 = n0 + 16;
    v8f acc[4][2] = {};
    for (int ks = 0; ks < K; ks += 32) {
        Frag bf0, bf1;
        const _Float16* bp0 = Wt + (size_t)n0 * K + ks + hsel * 16;
        bf0.u[0] = *(const u32x4*)(bp0);
        bf0.u[1] = *(const u32x4*)(bp0 + 8);
        const _Float16* bp1 = Wt + (size_t)n1 * K + ks + hsel * 16;
        bf1.u[0] = *(const u32x4*)(bp1);
        bf1.u[1] = *(const u32x4*)(bp1 + 8);
        #pragma unroll
        for (int mt = 0; mt < 4; ++mt) {
            Frag a;
            const _Float16* ap = src + (mt * 16 + l16) * srcStride + ks + hsel * 16;
            a.u[0] = *(const u32x4*)(ap);
            a.u[1] = *(const u32x4*)(ap + 8);
            acc[mt][0] = __builtin_amdgcn_wmma_f32_16x16x32_f16(
                false, a.h, false, bf0.h, (short)0, acc[mt][0], false, false);
            acc[mt][1] = __builtin_amdgcn_wmma_f32_16x16x32_f16(
                false, a.h, false, bf1.h, (short)0, acc[mt][1], false, false);
        }
    }
    float bv0 = bias[n0], bv1 = bias[n1];
    #pragma unroll
    for (int mt = 0; mt < 4; ++mt)
        #pragma unroll
        for (int i = 0; i < 8; ++i) {
            int m = mt * 16 + i + hsel * 8;      // C/D layout: M = i + (lane<16?0:8)
            float v0 = acc[mt][0][i] + bv0;
            float v1 = acc[mt][1][i] + bv1;
            if (relu) { v0 = fmaxf(v0, 0.f); v1 = fmaxf(v1, 0.f); }
            dst[m * dstStride + n0] = (_Float16)v0;
            dst[m * dstStride + n1] = (_Float16)v1;
        }
}

// ---------------- fused per-corner MLP kernel ----------------
__global__ __launch_bounds__(256) void liif_mlp(
    const _Float16* __restrict__ featH,       // [B][64][64][64] f16
    const float* __restrict__ coord,          // [B*Q][2]
    const float* __restrict__ cell,           // [B*Q][2]
    const _Float16* __restrict__ Wt0,         // [256][608]
    const _Float16* __restrict__ Wt1,         // [256][256]
    const _Float16* __restrict__ Wt2,
    const _Float16* __restrict__ Wt3,
    const float* __restrict__ b0, const float* __restrict__ b1,
    const float* __restrict__ b2, const float* __restrict__ b3,
    const float* __restrict__ w4,             // [256][3] f32
    const float* __restrict__ b4,             // [3]
    float* __restrict__ preds,                // [4*BQ][3]
    float* __restrict__ areas)                // [4*BQ]
{
    __shared__ __align__(16) _Float16 X [LIIF_MT * LIIF_KIN];
    __shared__ __align__(16) _Float16 H1[LIIF_MT * LIIF_HID];
    __shared__ __align__(16) _Float16 H2[LIIF_MT * LIIF_HID];
    __shared__ int   mB[LIIF_MT], mIY[LIIF_MT], mIX[LIIF_MT];
    __shared__ float mR0[LIIF_MT], mR1[LIIF_MT], mC0[LIIF_MT], mC1[LIIF_MT], mArea[LIIF_MT];

    const int tid  = threadIdx.x;
    const int lane = tid & 31;
    const int wave = tid >> 5;
    const long rowBase = (long)blockIdx.x * LIIF_MT;

    // ---- per-row corner metadata ----
    if (tid < LIIF_MT) {
        long rowIdx = rowBase + tid;
        int t = (int)(rowIdx / LIIF_BQ);     // corner 0..3 : (vx,vy) = ((t&2)?+1:-1, (t&1)?+1:-1)
        int s = (int)(rowIdx % LIIF_BQ);
        int b = s >> 13;                     // s / Q
        float c0  = coord[2 * s], c1 = coord[2 * s + 1];
        float ce0 = cell[2 * s],  ce1 = cell[2 * s + 1];
        float vx = (t & 2) ? 1.f : -1.f;
        float vy = (t & 1) ? 1.f : -1.f;
        const float r = 1.0f / 64.0f;
        float cy = fminf(fmaxf(c0 + vx * r + 1e-6f, -1.f + 1e-6f), 1.f - 1e-6f);
        float cx = fminf(fmaxf(c1 + vy * r + 1e-6f, -1.f + 1e-6f), 1.f - 1e-6f);
        int iy = (int)rintf(((cy + 1.f) * 64.f - 1.f) * 0.5f);
        int ix = (int)rintf(((cx + 1.f) * 64.f - 1.f) * 0.5f);
        iy = min(max(iy, 0), 63);
        ix = min(max(ix, 0), 63);
        float qy = -1.f + (2.f * iy + 1.f) * r;
        float qx = -1.f + (2.f * ix + 1.f) * r;
        float rel0 = (c0 - qy) * 64.f;
        float rel1 = (c1 - qx) * 64.f;
        mB[tid] = b; mIY[tid] = iy; mIX[tid] = ix;
        mR0[tid] = rel0; mR1[tid] = rel1;
        mC0[tid] = ce0 * 64.f; mC1[tid] = ce1 * 64.f;
        mArea[tid] = fabsf(rel0 * rel1) + 1e-9f;
    }
    __syncthreads();

    // ---- build X[m][k']: k' = p*64 + c (contiguous 64-ch runs per 3x3 pixel), then rel/cell, then zero pad
    for (int d = tid; d < LIIF_MT * (LIIF_KIN / 2); d += 256) {
        int m  = d / (LIIF_KIN / 2);
        int k  = (d % (LIIF_KIN / 2)) * 2;
        unsigned int val = 0;
        if (k < 576) {
            int p = k >> 6, c = k & 63;
            int yy = mIY[m] + (p / 3) - 1;
            int xx = mIX[m] + (p % 3) - 1;
            if (yy >= 0 && yy < 64 && xx >= 0 && xx < 64)
                val = *(const unsigned int*)&featH[(((mB[m] * 64 + yy) * 64 + xx) * 64) + c];
        } else if (k == 576) {
            union { unsigned int u; _Float16 h[2]; } pk;
            pk.h[0] = (_Float16)mR0[m]; pk.h[1] = (_Float16)mR1[m];
            val = pk.u;
        } else if (k == 578) {
            union { unsigned int u; _Float16 h[2]; } pk;
            pk.h[0] = (_Float16)mC0[m]; pk.h[1] = (_Float16)mC1[m];
            val = pk.u;
        }
        *(unsigned int*)&X[m * LIIF_KIN + k] = val;
    }
    __syncthreads();

    // ---- 4 WMMA layers fused in LDS ----
    mlp_layer(X,  LIIF_KIN, LIIF_KIN, Wt0, b0, H1, LIIF_HID, true, lane, wave);
    __syncthreads();
    mlp_layer(H1, LIIF_HID, LIIF_HID, Wt1, b1, H2, LIIF_HID, true, lane, wave);
    __syncthreads();
    mlp_layer(H2, LIIF_HID, LIIF_HID, Wt2, b2, H1, LIIF_HID, true, lane, wave);
    __syncthreads();
    mlp_layer(H1, LIIF_HID, LIIF_HID, Wt3, b3, H2, LIIF_HID, true, lane, wave);
    __syncthreads();

    // ---- final 256 -> 3 layer (VALU, negligible FLOPs) ----
    if (tid < LIIF_MT * 3) {
        int m = tid / 3, o = tid % 3;
        float s = b4[o];
        for (int k = 0; k < LIIF_HID; ++k)
            s += (float)H2[m * LIIF_HID + k] * w4[k * 3 + o];
        preds[(rowBase + m) * 3 + o] = s;
    }
    if (tid < LIIF_MT) areas[rowBase + tid] = mArea[tid];
}

// ---------------- 4-corner area-weighted blend ----------------
__global__ __launch_bounds__(256) void liif_reduce(
    const float* __restrict__ preds, const float* __restrict__ areas,
    float* __restrict__ out)
{
    int s = blockIdx.x * 256 + threadIdx.x;
    if (s >= LIIF_BQ) return;
    float a0 = areas[0 * LIIF_BQ + s];
    float a1 = areas[1 * LIIF_BQ + s];
    float a2 = areas[2 * LIIF_BQ + s];
    float a3 = areas[3 * LIIF_BQ + s];
    float tot = a0 + a1 + a2 + a3;
    float q0 = a3 / tot, q1 = a2 / tot, q2 = a1 / tot, q3 = a0 / tot;
    #pragma unroll
    for (int o = 0; o < 3; ++o) {
        out[s * 3 + o] =
            preds[((size_t)0 * LIIF_BQ + s) * 3 + o] * q0 +
            preds[((size_t)1 * LIIF_BQ + s) * 3 + o] * q1 +
            preds[((size_t)2 * LIIF_BQ + s) * 3 + o] * q2 +
            preds[((size_t)3 * LIIF_BQ + s) * 3 + o] * q3;
    }
}

// ---------------- host launch ----------------
extern "C" void kernel_launch(void* const* d_in, const int* in_sizes, int n_in,
                              void* d_out, int out_size, void* d_ws, size_t ws_size,
                              hipStream_t stream) {
    (void)in_sizes; (void)n_in; (void)out_size; (void)ws_size;
    const float* inp   = (const float*)d_in[0];
    const float* coord = (const float*)d_in[1];
    const float* cell  = (const float*)d_in[2];
    const float* enc_w = (const float*)d_in[3];
    const float* enc_b = (const float*)d_in[4];
    const float* w0 = (const float*)d_in[5];
    const float* b0 = (const float*)d_in[6];
    const float* w1 = (const float*)d_in[7];
    const float* b1 = (const float*)d_in[8];
    const float* w2 = (const float*)d_in[9];
    const float* b2 = (const float*)d_in[10];
    const float* w3 = (const float*)d_in[11];
    const float* b3 = (const float*)d_in[12];
    const float* w4 = (const float*)d_in[13];
    const float* b4 = (const float*)d_in[14];
    float* out = (float*)d_out;

    // workspace layout (all 16B-aligned offsets)
    char* ws = (char*)d_ws;
    constexpr size_t offFeat  = 0;                                            // 8*64*64*64*2 = 4,194,304
    constexpr size_t offWt0   = offFeat + (size_t)LIIF_B * 64 * 64 * 64 * 2;  // 256*608*2    =   311,296
    constexpr size_t offWt1   = offWt0  + (size_t)LIIF_HID * LIIF_KIN * 2;
    constexpr size_t offWt2   = offWt1  + (size_t)LIIF_HID * LIIF_HID * 2;
    constexpr size_t offWt3   = offWt2  + (size_t)LIIF_HID * LIIF_HID * 2;
    constexpr size_t offPred  = offWt3  + (size_t)LIIF_HID * LIIF_HID * 2;    // 4*BQ*3*4 = 3,145,728
    constexpr size_t offArea  = offPred + (size_t)4 * LIIF_BQ * 3 * 4;        // 4*BQ*4   = 1,048,576

    _Float16* featH = (_Float16*)(ws + offFeat);
    _Float16* Wt0   = (_Float16*)(ws + offWt0);
    _Float16* Wt1   = (_Float16*)(ws + offWt1);
    _Float16* Wt2   = (_Float16*)(ws + offWt2);
    _Float16* Wt3   = (_Float16*)(ws + offWt3);
    float*    preds = (float*)(ws + offPred);
    float*    areas = (float*)(ws + offArea);

    liif_conv<<<(LIIF_B * 64 * 64 * 64) / 256, 256, 0, stream>>>(inp, enc_w, enc_b, featH);
    liif_prep_wt0<<<(LIIF_HID * LIIF_KIN + 255) / 256, 256, 0, stream>>>(w0, Wt0);
    liif_prep_wth<<<(LIIF_HID * LIIF_HID) / 256, 256, 0, stream>>>(w1, Wt1);
    liif_prep_wth<<<(LIIF_HID * LIIF_HID) / 256, 256, 0, stream>>>(w2, Wt2);
    liif_prep_wth<<<(LIIF_HID * LIIF_HID) / 256, 256, 0, stream>>>(w3, Wt3);

    liif_mlp<<<(4 * LIIF_BQ) / LIIF_MT, 256, 0, stream>>>(
        featH, coord, cell, Wt0, Wt1, Wt2, Wt3,
        b0, b1, b2, b3, w4, b4, preds, areas);

    liif_reduce<<<LIIF_BQ / 256, 256, 0, stream>>>(preds, areas, out);
}
